// DKN_2680059593426
// MI455X (gfx1250) — matrix operand order repaired and checked by reading, not compiled, not run on hardware
//
#include <hip/hip_runtime.h>
#include <cstdint>
#include <cstddef>

// ---------------------------------------------------------------------------
// CDNA5 (gfx1250) wave32 WMMA types
// ---------------------------------------------------------------------------
typedef __attribute__((ext_vector_type(16))) __bf16          v16bf;
typedef __attribute__((ext_vector_type(16))) unsigned short  u16x16;
typedef __attribute__((ext_vector_type(8)))  float           v8f;
typedef unsigned short ushort_t;
typedef unsigned long long u64;

__device__ __forceinline__ unsigned short f2bf(float f) {
  unsigned int u = __float_as_uint(f);
  unsigned int r = u + 0x7FFFu + ((u >> 16) & 1u);
  return (unsigned short)(r >> 16);
}
__device__ __forceinline__ int imin(int a, int b) { return a < b ? a : b; }

// ---------------------------------------------------------------------------
// Zero-pad to NHWC [306][308][4] bf16 (pad=25; ghost cols/channels zero)
// ---------------------------------------------------------------------------
__global__ void pad_to_bf16_nhwc(const float* __restrict__ src,
                                 ushort_t* __restrict__ dst, int realC) {
  const int total = 306 * 308 * 4;
  int idx = blockIdx.x * blockDim.x + threadIdx.x;
  if (idx >= total) return;
  int c = idx & 3;
  int x = (idx >> 2) % 308;
  int y = idx / (308 * 4);
  int sy = y - 25, sx = x - 25;
  ushort_t v = 0;
  if (c < realC && (unsigned)sy < 256u && (unsigned)sx < 256u)
    v = f2bf(src[(size_t)c * 65536 + (size_t)sy * 256 + sx]);
  dst[idx] = v;
}

// ---------------------------------------------------------------------------
// Weight packing: OIHW f32 -> [Cout][ky][kx][Cin] bf16 (k-order = ky,kx,c)
// ---------------------------------------------------------------------------
__global__ void pack_w(const float* __restrict__ src, ushort_t* __restrict__ dst,
                       int Cout, int C, int K) {
  int total = Cout * K * K * C;
  int idx = blockIdx.x * blockDim.x + threadIdx.x;
  if (idx >= total) return;
  int c = idx % C;
  int t = idx / C;
  int kx = t % K; t /= K;
  int ky = t % K;
  int n  = t / K;
  dst[idx] = f2bf(src[(((size_t)n * C + c) * K + ky) * K + kx]);
}

// conv1 pack: [32][Cin][7][7] f32 -> [32][7][8][4] bf16; kx==7 / c>=Cin zero.
__global__ void pack_w1(const float* __restrict__ src, ushort_t* __restrict__ dst,
                        int Cin) {
  const int total = 32 * 224;
  int idx = blockIdx.x * blockDim.x + threadIdx.x;
  if (idx >= total) return;
  int c  = idx & 3;
  int kx = (idx >> 2) & 7;
  int ky = (idx >> 5) % 7;
  int n  = idx / 224;
  ushort_t v = 0;
  if (kx < 7 && c < Cin)
    v = f2bf(src[(((size_t)n * Cin + c) * 7 + ky) * 7 + kx]);
  dst[idx] = v;
}

// ---------------------------------------------------------------------------
// Implicit-GEMM conv, NHWC bf16, packed bf16 weights, 2x2 register tiling:
// one wave computes a 32(M=spatial) x 32(N=Cout) tile with 4 f32 accumulators,
// reusing each A/B fragment twice -> 8x b128-class loads per 4 WMMAs.
// gridDim.y = batch (shift pass); shiftMode: conv2 derives (oy0,ox0) from it.
// K-loop (ky,kx,cb32): channels innermost -> contiguous fragment loads,
// no divisions; M/N edges clamped (stores bounds-checked), EXEC uniform.
// Epilogue fuses bias + eval-BatchNorm + relu/sigmoid.
// ---------------------------------------------------------------------------
__global__ __launch_bounds__(128)
void conv_bf16_wmma(const ushort_t* __restrict__ in, int inSliceU,
                    int rowStrideU, int xStrideU,
                    int oy0, int ox0, int shiftMode, int stride,
                    const ushort_t* __restrict__ wpack,
                    const float* __restrict__ bias,
                    const float* __restrict__ bg, const float* __restrict__ bbe,
                    const float* __restrict__ bm, const float* __restrict__ bv,
                    int kyCount, int kxCount, int chunkU, int Ktot,
                    int Cout, int OW, int act,
                    ushort_t* __restrict__ outb, int outSliceU,
                    float* __restrict__ outf, int outSliceF) {
  const int lane = threadIdx.x & 31;
  const int wave = (blockIdx.x * blockDim.x + threadIdx.x) >> 5;
  const int M  = OW * OW;
  const int mt = (M + 31) >> 5;
  const int nt = (Cout + 31) >> 5;
  if (wave >= mt * nt) return;            // uniform per wave
  const int mtile = wave % mt;
  const int ntile = wave / mt;
  const int l15   = lane & 15;
  const int lhalf = lane >> 4;

  const int b = blockIdx.y;
  if (shiftMode) { oy0 = b >> 2; ox0 = b & 3; }
  const ushort_t* inB = in + (size_t)b * inSliceU;

  // two A-row blocks (spatial), clamped
  const int mr0 = imin(mtile * 32 + l15, M - 1);
  const int mr1 = imin(mtile * 32 + 16 + l15, M - 1);
  int oy0_ = mr0 / OW, ox0_ = mr0 - oy0_ * OW;
  int oy1_ = mr1 / OW, ox1_ = mr1 - oy1_ * OW;
  const ushort_t* ab0 = inB + (size_t)(oy0_ * stride + oy0) * rowStrideU +
                              (size_t)(ox0_ * stride + ox0) * xStrideU + lhalf * 8;
  const ushort_t* ab1 = inB + (size_t)(oy1_ * stride + oy0) * rowStrideU +
                              (size_t)(ox1_ * stride + ox0) * xStrideU + lhalf * 8;

  // two B-column blocks (output channels), clamped for loads
  const int nc0 = ntile * 32 + l15;
  const int nc1 = ntile * 32 + 16 + l15;
  const ushort_t* bp0 = wpack + (size_t)imin(nc0, Cout - 1) * Ktot + lhalf * 8;
  const ushort_t* bp1 = wpack + (size_t)imin(nc1, Cout - 1) * Ktot + lhalf * 8;

  union AFrag { u16x16 v; u64 q[4]; };
  union BFrag { u16x16 v; uint4 d[2]; };

  v8f acc00 = {0.f,0.f,0.f,0.f,0.f,0.f,0.f,0.f};
  v8f acc01 = acc00, acc10 = acc00, acc11 = acc00;

  for (int ky = 0; ky < kyCount; ++ky) {
    const ushort_t* ar0 = ab0 + (size_t)ky * rowStrideU;
    const ushort_t* ar1 = ab1 + (size_t)ky * rowStrideU;
    for (int kx = 0; kx < kxCount; ++kx) {
      const ushort_t* ap0 = ar0 + (size_t)kx * xStrideU;
      const ushort_t* ap1 = ar1 + (size_t)kx * xStrideU;
      for (int cb = 0; cb < chunkU; cb += 32) {
        __builtin_prefetch(bp0 + 96, 0, 1);          // -> global_prefetch_b8
        AFrag a0, a1; BFrag b0, b1;
        a0.q[0] = *(const u64*)(ap0 + cb);
        a0.q[1] = *(const u64*)(ap0 + cb + 4);
        a0.q[2] = *(const u64*)(ap0 + cb + 16);
        a0.q[3] = *(const u64*)(ap0 + cb + 20);
        a1.q[0] = *(const u64*)(ap1 + cb);
        a1.q[1] = *(const u64*)(ap1 + cb + 4);
        a1.q[2] = *(const u64*)(ap1 + cb + 16);
        a1.q[3] = *(const u64*)(ap1 + cb + 20);
        b0.d[0] = *(const uint4*)(bp0);
        b0.d[1] = *(const uint4*)(bp0 + 16);
        b1.d[0] = *(const uint4*)(bp1);
        b1.d[1] = *(const uint4*)(bp1 + 16);
        bp0 += 32; bp1 += 32;
        acc00 = __builtin_amdgcn_wmma_f32_16x16x32_bf16(
            false, a0.v, false, b0.v, (short)0, acc00, false, false);
        acc01 = __builtin_amdgcn_wmma_f32_16x16x32_bf16(
            false, a0.v, false, b1.v, (short)0, acc01, false, false);
        acc10 = __builtin_amdgcn_wmma_f32_16x16x32_bf16(
            false, a1.v, false, b0.v, (short)0, acc10, false, false);
        acc11 = __builtin_amdgcn_wmma_f32_16x16x32_bf16(
            false, a1.v, false, b1.v, (short)0, acc11, false, false);
      }
    }
  }

  ushort_t* outbB = outb ? outb + (size_t)b * outSliceU : (ushort_t*)nullptr;
  float*    outfB = outf ? outf + (size_t)b * outSliceF : (float*)nullptr;

#pragma unroll
  for (int j = 0; j < 2; ++j) {
    int nc = ntile * 32 + j * 16 + l15;
    if (nc >= Cout) continue;
    float s = 1.f, t = 0.f;
    const float bb = bias[nc];
    if (bg) {                                // folded eval-mode BatchNorm
      float iv = rsqrtf(bv[nc] + 1e-5f);
      s = bg[nc] * iv;
      t = bbe[nc] - bm[nc] * s;
    }
#pragma unroll
    for (int th = 0; th < 2; ++th) {
      v8f accv = (th == 0) ? (j == 0 ? acc00 : acc01)
                           : (j == 0 ? acc10 : acc11);
#pragma unroll
      for (int i2 = 0; i2 < 8; ++i2) {
        int mo = mtile * 32 + th * 16 + i2 + lhalf * 8;   // C/D layout
        if (mo >= M) continue;
        float val = (accv[i2] + bb) * s + t;
        if (act == 1)      val = fmaxf(val, 0.f);
        else if (act == 2) val = 1.f / (1.f + expf(-val));
        if (outfB) outfB[(size_t)nc * M + mo] = val;          // heads [C][M]
        else       outbB[(size_t)mo * Cout + nc] = f2bf(val); // NHWC bf16
      }
    }
  }
}

// ---------------------------------------------------------------------------
// Combine image/depth heads and scatter with stride 4 (optionally batched)
// ---------------------------------------------------------------------------
__global__ void stitch(const float* __restrict__ wi, const float* __restrict__ oi,
                       const float* __restrict__ wd, const float* __restrict__ od,
                       float* __restrict__ Wb, float* __restrict__ Ob,
                       int si, int sj, int sliceMode) {
  int p = blockIdx.x * blockDim.x + threadIdx.x;
  if (p >= 64 * 64) return;
  int bb = blockIdx.y;
  if (sliceMode) {
    si = bb >> 2; sj = bb & 3;
    wi += (size_t)bb * 9 * 4096;  oi += (size_t)bb * 18 * 4096;
    wd += (size_t)bb * 9 * 4096;  od += (size_t)bb * 18 * 4096;
  }
  int y = p >> 6, x = p & 63;
  int dp = (si + 4 * y) * 256 + (sj + 4 * x);
  float ww[9], mean = 0.f;
#pragma unroll
  for (int c = 0; c < 9; ++c) {
    ww[c] = wi[c * 4096 + p] * wd[c * 4096 + p];
    mean += ww[c];
  }
  mean *= (1.f / 9.f);
#pragma unroll
  for (int c = 0; c < 9; ++c)  Wb[c * 65536 + dp] = ww[c] - mean;
#pragma unroll
  for (int c = 0; c < 18; ++c) Ob[c * 65536 + dp] = oi[c * 4096 + p] * od[c * 4096 + p];
}

// ---------------------------------------------------------------------------
// Fused unfold(15,pad=7) + bilinear grid_sample(zeros) + 3x3 weighted sum
// + residual. Patch-space sample position = grid + offset - 0.5.
// ---------------------------------------------------------------------------
__global__ void dkn_sample(const float* __restrict__ depth,
                           const float* __restrict__ Wb,
                           const float* __restrict__ Ob,
                           float* __restrict__ out) {
  int p = blockIdx.x * blockDim.x + threadIdx.x;
  if (p >= 65536) return;
  int y = p >> 8, x = p & 255;
  float acc = 0.f;
#pragma unroll
  for (int a = 0; a < 3; ++a) {
#pragma unroll
    for (int b = 0; b < 3; ++b) {
      int tc = a * 3 + b;
      float offx = Ob[(2 * tc + 0) * 65536 + p];
      float offy = Ob[(2 * tc + 1) * 65536 + p];
      float xf = (float)(7 + a) + offx - 0.5f;
      float yf = (float)(7 + b) + offy - 0.5f;
      float x0f = floorf(xf), y0f = floorf(yf);
      int x0 = (int)x0f, y0 = (int)y0f;
      float fx = xf - x0f, fy = yf - y0f;
      auto tap = [&](int yi, int xi) -> float {
        if ((unsigned)yi >= 15u || (unsigned)xi >= 15u) return 0.f;
        int iy = y - 7 + yi, ix = x - 7 + xi;
        if ((unsigned)iy >= 256u || (unsigned)ix >= 256u) return 0.f;
        return depth[iy * 256 + ix];
      };
      float v00 = tap(y0, x0),     v01 = tap(y0, x0 + 1);
      float v10 = tap(y0 + 1, x0), v11 = tap(y0 + 1, x0 + 1);
      float sv = (1.f - fx) * (1.f - fy) * v00 + fx * (1.f - fy) * v01 +
                 (1.f - fx) * fy * v10        + fx * fy * v11;
      acc += sv * Wb[tc * 65536 + p];
    }
  }
  out[p] = acc + depth[p];
}

// ---------------------------------------------------------------------------
// Host orchestration
// ---------------------------------------------------------------------------
extern "C" void kernel_launch(void* const* d_in, const int* in_sizes, int n_in,
                              void* d_out, int out_size, void* d_ws, size_t ws_size,
                              hipStream_t stream) {
  (void)n_in; (void)out_size;

  const float *image, *depth;
  if (in_sizes[0] == 3 * 256 * 256) { image = (const float*)d_in[0]; depth = (const float*)d_in[1]; }
  else                              { depth = (const float*)d_in[0]; image = (const float*)d_in[1]; }

  auto has4704 = [&](int base) {
    for (int t = 0; t < 30; ++t) if (in_sizes[base + t] == 4704) return true;
    return false;
  };
  int imgBase = has4704(2) ? 2 : 32;
  int depBase = (imgBase == 2) ? 32 : 2;

  struct Net { const float *w[9], *b[9], *bn[3][4]; };
  auto mapNet = [&](int base) {
    Net n;
    bool sorted = in_sizes[base] < 1000;   // insertion order starts with w1
    static const int wIns[9] = {0,1,2,3,4,5,6,7,8};
    static const int bIns[9] = {9,10,11,12,13,14,15,16,17};
    static const int bnIns[3][4] = {{18,19,20,21},{22,23,24,25},{26,27,28,29}};
    static const int wSrt[9] = {21,22,23,24,25,26,27,29,28};
    static const int bSrt[9] = {0,1,2,3,4,5,6,11,10};
    static const int bnSrt[3][4] = {{12,7,15,18},{13,8,16,19},{14,9,17,20}};
    for (int q = 0; q < 9; ++q) {
      n.w[q] = (const float*)d_in[base + (sorted ? wSrt[q] : wIns[q])];
      n.b[q] = (const float*)d_in[base + (sorted ? bSrt[q] : bIns[q])];
    }
    for (int q = 0; q < 3; ++q)
      for (int r2 = 0; r2 < 4; ++r2)
        n.bn[q][r2] = (const float*)d_in[base + (sorted ? bnSrt[q][r2] : bnIns[q][r2])];
    return n;
  };
  Net nImg = mapNet(imgBase);
  Net nDep = mapNet(depBase);

  // Batch all 16 shift passes into gridDim.y if workspace permits (~102 MB),
  // else run them sequentially (~27 MB). Deterministic given fixed ws_size.
  const bool batched = ws_size >= ((size_t)110 << 20);
  const int  NB = batched ? 16 : 1;
  const int  SL_A = 148 * 148 * 32;   // per-pass ping slice (ushorts)
  const int  SL_B = 144 * 144 * 64;   // per-pass pong slice (ushorts)

  char* wsp = (char*)d_ws;
  auto carve = [&](size_t bytes) {
    void* p = (void*)wsp;
    wsp += (bytes + 255) & ~(size_t)255;
    return p;
  };
  ushort_t* padImg = (ushort_t*)carve((size_t)306 * 308 * 4 * 2);
  ushort_t* padDep = (ushort_t*)carve((size_t)306 * 308 * 4 * 2);
  ushort_t* c1Img  = (ushort_t*)carve((size_t)300 * 300 * 32 * 2);
  ushort_t* c1Dep  = (ushort_t*)carve((size_t)300 * 300 * 32 * 2);
  ushort_t* bufA   = (ushort_t*)carve((size_t)NB * SL_A * 2);
  ushort_t* bufB   = (ushort_t*)carve((size_t)NB * SL_B * 2);
  float* wiBuf = (float*)carve((size_t)NB * 9  * 4096 * 4);
  float* oiBuf = (float*)carve((size_t)NB * 18 * 4096 * 4);
  float* wdBuf = (float*)carve((size_t)NB * 9  * 4096 * 4);
  float* odBuf = (float*)carve((size_t)NB * 18 * 4096 * 4);
  float* Wbuf  = (float*)carve((size_t)9  * 65536 * 4);
  float* Obuf  = (float*)carve((size_t)18 * 65536 * 4);

  static const int pkElems[9] = {32 * 224, 32 * 128, 64 * 800, 64 * 256,
                                 128 * 1600, 128 * 1152, 128 * 1152,
                                 9 * 128, 18 * 128};
  ushort_t* pk[2][9];
  for (int nn = 0; nn < 2; ++nn)
    for (int q = 0; q < 9; ++q)
      pk[nn][q] = (ushort_t*)carve((size_t)pkElems[q] * 2);

  static const int lyC[9]  = {0, 32, 32, 64, 64, 128, 128, 128, 128};
  static const int lyK[9]  = {7, 2, 5, 2, 5, 3, 3, 1, 1};
  static const int lyCo[9] = {32, 32, 64, 64, 128, 128, 128, 9, 18};
  const Net* nets[2] = {&nImg, &nDep};
  for (int nn = 0; nn < 2; ++nn) {
    pack_w1<<<(32 * 224 + 255) / 256, 256, 0, stream>>>(
        nets[nn]->w[0], pk[nn][0], nn == 0 ? 3 : 1);
    for (int q = 1; q < 9; ++q) {
      int tot = lyCo[q] * lyK[q] * lyK[q] * lyC[q];
      pack_w<<<(tot + 255) / 256, 256, 0, stream>>>(
          nets[nn]->w[q], pk[nn][q], lyCo[q], lyC[q], lyK[q]);
    }
  }

  auto conv = [&](const ushort_t* in, int inSliceU, int rowStrideU, int xStrideU,
                  int oy0, int ox0, int shiftMode, int stride,
                  const ushort_t* wp, const float* bvec, const float* const* bn,
                  int kyCount, int kxCount, int chunkU, int Ktot,
                  int Cout, int OW, int act,
                  ushort_t* ob, int obSlice, float* of, int ofSlice, int nb) {
    int M = OW * OW;
    int waves = ((M + 31) / 32) * ((Cout + 31) / 32);
    int blocks = (waves + 3) / 4;   // 4 wave32 per 128-thread block
    conv_bf16_wmma<<<dim3(blocks, nb), 128, 0, stream>>>(
        in, inSliceU, rowStrideU, xStrideU, oy0, ox0, shiftMode, stride,
        wp, bvec,
        bn ? bn[0] : nullptr, bn ? bn[1] : nullptr,
        bn ? bn[2] : nullptr, bn ? bn[3] : nullptr,
        kyCount, kxCount, chunkU, Ktot, Cout, OW, act,
        ob, obSlice, of, ofSlice);
  };

  pad_to_bf16_nhwc<<<(306 * 308 * 4 + 255) / 256, 256, 0, stream>>>(image, padImg, 3);
  pad_to_bf16_nhwc<<<(306 * 308 * 4 + 255) / 256, 256, 0, stream>>>(depth, padDep, 1);

  // conv1 ONCE per net, dense 300x300 (shift passes become window offsets)
  conv(padImg, 0, 308 * 4, 4, 0, 0, 0, 1, pk[0][0], nImg.b[0], nImg.bn[0],
       7, 1, 32, 224, 32, 300, 1, c1Img, 0, nullptr, 0, 1);
  conv(padDep, 0, 308 * 4, 4, 0, 0, 0, 1, pk[1][0], nDep.b[0], nDep.bn[0],
       7, 1, 32, 224, 32, 300, 1, c1Dep, 0, nullptr, 0, 1);

  auto runTail = [&](int nn, const Net& n, const ushort_t* c1,
                     int si, int sj, int shiftMode, int nb,
                     float* wOut, float* oOut) {
    conv(c1,   0,    300 * 32, 32, si, sj, shiftMode, 2, pk[nn][1], n.b[1], nullptr,
         2, 2, 32, 128, 32, 148, 1, bufA, SL_A, nullptr, 0, nb);
    conv(bufA, SL_A, 148 * 32, 32, 0, 0, 0, 1, pk[nn][2], n.b[2], n.bn[1],
         5, 5, 32, 800, 64, 144, 1, bufB, SL_B, nullptr, 0, nb);
    conv(bufB, SL_B, 144 * 64, 64, 0, 0, 0, 2, pk[nn][3], n.b[3], nullptr,
         2, 2, 64, 256, 64, 72, 1, bufA, SL_A, nullptr, 0, nb);
    conv(bufA, SL_A, 72 * 64,  64, 0, 0, 0, 1, pk[nn][4], n.b[4], n.bn[2],
         5, 5, 64, 1600, 128, 68, 1, bufB, SL_B, nullptr, 0, nb);
    conv(bufB, SL_B, 68 * 128, 128, 0, 0, 0, 1, pk[nn][5], n.b[5], nullptr,
         3, 3, 128, 1152, 128, 66, 1, bufA, SL_A, nullptr, 0, nb);
    conv(bufA, SL_A, 66 * 128, 128, 0, 0, 0, 1, pk[nn][6], n.b[6], nullptr,
         3, 3, 128, 1152, 128, 64, 1, bufB, SL_B, nullptr, 0, nb);
    conv(bufB, SL_B, 64 * 128, 128, 0, 0, 0, 1, pk[nn][7], n.b[7], nullptr,
         1, 1, 128, 128, 9, 64, 2, nullptr, 0, wOut, 9 * 4096, nb);   // sigmoid
    conv(bufB, SL_B, 64 * 128, 128, 0, 0, 0, 1, pk[nn][8], n.b[8], nullptr,
         1, 1, 128, 128, 18, 64, 0, nullptr, 0, oOut, 18 * 4096, nb); // linear
  };

  if (batched) {
    runTail(0, nImg, c1Img, 0, 0, /*shiftMode=*/1, 16, wiBuf, oiBuf);
    runTail(1, nDep, c1Dep, 0, 0, /*shiftMode=*/1, 16, wdBuf, odBuf);
    stitch<<<dim3(16, 16), 256, 0, stream>>>(wiBuf, oiBuf, wdBuf, odBuf,
                                             Wbuf, Obuf, 0, 0, 1);
  } else {
    for (int si = 0; si < 4; ++si)
      for (int sj = 0; sj < 4; ++sj) {
        runTail(0, nImg, c1Img, si, sj, 0, 1, wiBuf, oiBuf);
        runTail(1, nDep, c1Dep, si, sj, 0, 1, wdBuf, odBuf);
        stitch<<<dim3(16, 1), 256, 0, stream>>>(wiBuf, oiBuf, wdBuf, odBuf,
                                                Wbuf, Obuf, si, sj, 0);
      }
  }

  dkn_sample<<<256, 256, 0, stream>>>(depth, Wbuf, Obuf, (float*)d_out);
}
